// TopBDNet_80848464380047
// MI455X (gfx1250) — matrix operand order repaired and checked by reading, not compiled, not run on hardware
//
#include <hip/hip_runtime.h>
#include <hip/hip_bf16.h>

// ---------------------------------------------------------------------------
// Types for CDNA5 WMMA (wave32, 16x16x32 bf16 -> f32 accumulate)
// ---------------------------------------------------------------------------
typedef __attribute__((ext_vector_type(16))) __bf16 v16bf;
typedef __attribute__((ext_vector_type(8)))  float  v8f;

union FragBf {
    v16bf  v;
    float4 f4[2];   // two 16-byte chunks
    __bf16 h[16];
};

#define EPS 1e-5f

// ---------------------------------------------------------------------------
// f32 -> bf16 elementwise convert
// ---------------------------------------------------------------------------
__global__ void cvt_bf16_kernel(const float* __restrict__ src,
                                __bf16* __restrict__ dst, int n) {
    int i = blockIdx.x * blockDim.x + threadIdx.x;
    if (i < n) dst[i] = (__bf16)src[i];
}

// ---------------------------------------------------------------------------
// Adaptive avg-pool to 1x1: one wave (32 lanes) reduces one 192-float channel.
// 8 channels per 256-thread block. Fully coalesced 128B loads.
// ---------------------------------------------------------------------------
__global__ __launch_bounds__(256) void pool_kernel(const float* __restrict__ x,
                                                   float* __restrict__ pooled,
                                                   __bf16* __restrict__ pooledBf) {
    const int w = threadIdx.x >> 5, l = threadIdx.x & 31;
    const size_t cidx = (size_t)blockIdx.x * 8 + w;   // 0 .. B*C-1
    const float* p = x + cidx * 192;
    float s = p[l] + p[l + 32] + p[l + 64] + p[l + 96] + p[l + 128] + p[l + 160];
    for (int off = 16; off > 0; off >>= 1) s += __shfl_down(s, off, 32);
    if (l == 0) {
        float m = s * (1.0f / 192.0f);
        pooled[cidx]   = m;
        pooledBf[cidx] = (__bf16)m;
    }
}

// ---------------------------------------------------------------------------
// Per-sample 4-way k-means (Lloyd, 10 iters + final assign).
// One workgroup (256 thr = 8 waves) per sample. Assignment step computes
// S = X @ C^T via v_wmma_f32_16x16x32_bf16 with centers padded to 16 cols.
// argmin uses (|c_k|^2 - 2*S[:,k]) -- the xs term is constant per row.
// Emits masked pooled matrix PkAll[(k*256+b), c] in bf16 for the H1 GEMM.
// ---------------------------------------------------------------------------
__global__ __launch_bounds__(256) void kmeans_kernel(const float* __restrict__ x,
                                                     const float* __restrict__ pooled,
                                                     __bf16* __restrict__ PkAll) {
    __shared__ float  centF[4][192];
    __shared__ __bf16 centB[16][192];   // rows 4..15 are zero padding (B cols)
    __shared__ float  sumC[4][192];
    __shared__ float  cn2[4];
    __shared__ int    cnt[4];
    __shared__ unsigned char ids[2048];
    __shared__ float  sS[8][16][4];     // per-wave S tile, clusters 0..3

    const int tid = threadIdx.x;
    const int w = tid >> 5, l = tid & 31;
    const int h = l >> 4, n16 = l & 15;
    const int b = blockIdx.x;
    const float* Xb = x + (size_t)b * 2048 * 192;

    // deterministic quartile init: rows 0, 512, 1024, 1536
    for (int i = tid; i < 4 * 192; i += 256) {
        int k = i / 192, d = i - k * 192;
        centF[k][d] = Xb[(size_t)(k * 512) * 192 + d];
    }
    for (int i = tid; i < 16 * 192; i += 256) {
        int k = i / 192, d = i - k * 192;
        if (k >= 4) centB[k][d] = (__bf16)0.0f;
    }

    for (int it = 0; it <= 10; ++it) {
        __syncthreads();
        for (int i = tid; i < 4 * 192; i += 256) {
            int k = i / 192, d = i - k * 192;
            centB[k][d] = (__bf16)centF[k][d];
            sumC[k][d]  = 0.0f;
        }
        if (tid < 4) {
            float s = 0.f;
            for (int d = 0; d < 192; ++d) { float c = centF[tid][d]; s += c * c; }
            cn2[tid] = s;
            cnt[tid] = 0;
        }
        __syncthreads();

        // ---- assignment: each wave owns 16 channel-row tiles (uniform trip count)
        for (int t = w; t < 128; t += 8) {
            const int c0 = t * 16;
            v8f acc = {0.f, 0.f, 0.f, 0.f, 0.f, 0.f, 0.f, 0.f};
            const float* Xr = Xb + (size_t)(c0 + n16) * 192;
            for (int k0 = 0; k0 < 192; k0 += 32) {
                const int ka = k0 + 8 * h;           // A: K {ka..ka+7, ka+16..ka+23}
                float4 q0 = *(const float4*)(Xr + ka);
                float4 q1 = *(const float4*)(Xr + ka + 4);
                float4 q2 = *(const float4*)(Xr + ka + 16);
                float4 q3 = *(const float4*)(Xr + ka + 20);
                FragBf a, bb;
                a.h[0]=(__bf16)q0.x;  a.h[1]=(__bf16)q0.y;  a.h[2]=(__bf16)q0.z;  a.h[3]=(__bf16)q0.w;
                a.h[4]=(__bf16)q1.x;  a.h[5]=(__bf16)q1.y;  a.h[6]=(__bf16)q1.z;  a.h[7]=(__bf16)q1.w;
                a.h[8]=(__bf16)q2.x;  a.h[9]=(__bf16)q2.y;  a.h[10]=(__bf16)q2.z; a.h[11]=(__bf16)q2.w;
                a.h[12]=(__bf16)q3.x; a.h[13]=(__bf16)q3.y; a.h[14]=(__bf16)q3.z; a.h[15]=(__bf16)q3.w;
                const int kb = k0 + 8 * h;           // B: K {kb..kb+7, kb+16..kb+23}
                const float4* bp = (const float4*)&centB[n16][kb];
                bb.f4[0] = bp[0]; bb.f4[1] = bp[2];
                acc = __builtin_amdgcn_wmma_f32_16x16x32_bf16(
                        false, a.v, false, bb.v, (short)0, acc, false, false);
            }
            if (n16 < 4) {
                for (int v = 0; v < 8; ++v) sS[w][v + 8 * h][n16] = acc[v];
            }
            __syncthreads();
            if (l < 16) {
                int c = c0 + l;
                float d0 = cn2[0] - 2.f * sS[w][l][0];
                float d1 = cn2[1] - 2.f * sS[w][l][1];
                float d2 = cn2[2] - 2.f * sS[w][l][2];
                float d3 = cn2[3] - 2.f * sS[w][l][3];
                int best = 0; float bv = d0;
                if (d1 < bv) { bv = d1; best = 1; }
                if (d2 < bv) { bv = d2; best = 2; }
                if (d3 < bv) { bv = d3; best = 3; }
                ids[c] = (unsigned char)best;
                if (it < 10) atomicAdd(&cnt[best], 1);
            }
            __syncthreads();
        }
        if (it == 10) break;

        // ---- center update: lane d accumulates dim d over all 2048 channels
        if (tid < 192) {
            const int d = tid;
            float a0 = 0.f, a1 = 0.f, a2 = 0.f, a3 = 0.f;
            for (int c = 0; c < 2048; ++c) {
                float v = Xb[(size_t)c * 192 + d];   // coalesced across lanes
                int id = ids[c];
                a0 += (id == 0) ? v : 0.f;
                a1 += (id == 1) ? v : 0.f;
                a2 += (id == 2) ? v : 0.f;
                a3 += (id == 3) ? v : 0.f;
            }
            sumC[0][d] = a0; sumC[1][d] = a1; sumC[2][d] = a2; sumC[3][d] = a3;
        }
        __syncthreads();
        for (int i = tid; i < 4 * 192; i += 256) {
            int k = i / 192, d = i - k * 192;
            int c = cnt[k];
            if (c > 0) centF[k][d] = sumC[k][d] / (float)c;   // cnt==0 keeps old
        }
    }
    __syncthreads();

    // ---- Pk[k,b,c] = pooled[b,c] * (ids[c]==k), stored bf16 row (k*256+b)
    const float* pb = pooled + (size_t)b * 2048;
    for (int c = tid; c < 2048; c += 256) {
        int id = ids[c];
        float pv = pb[c];
        for (int k = 0; k < 4; ++k)
            PkAll[((size_t)(k * 256 + b)) * 2048 + c] = (__bf16)((k == id) ? pv : 0.f);
    }
}

// ---------------------------------------------------------------------------
// C[M,N] = A[M,K] @ W[N,K]^T  (bf16 in, f32 accumulate) with fused epilogue.
// Block = 256 thr = 8 waves as 2(M) x 4(N) grid of 16x16 WMMA tiles -> 32x64.
// Fragments loaded directly from global (operands are L2-resident).
//   mode 0: +bias, BN1, relu, BN2            -> out[m*N+n]          (global branch)
//   mode 1: +bias, BN1, relu                 -> out[m*N+n]          (H1)
//   mode 2: +bias, BN1, relu, BN2[k*128+n]   -> out[b*512 + k*128+n] (H2 + concat)
// ---------------------------------------------------------------------------
__global__ __launch_bounds__(256) void gemm_bf16_kernel(
        const __bf16* __restrict__ A, const __bf16* __restrict__ W,
        int M, int N, int K, int mode,
        const float* __restrict__ bias,
        const float* __restrict__ g1, const float* __restrict__ be1,
        const float* __restrict__ m1, const float* __restrict__ v1,
        const float* __restrict__ g2, const float* __restrict__ be2,
        const float* __restrict__ m2, const float* __restrict__ v2,
        __bf16* __restrict__ out) {
    const int tid = threadIdx.x;
    const int w = tid >> 5, l = tid & 31;
    const int h = l >> 4, nn = l & 15;
    const int wm = w >> 2, wn = w & 3;
    const int m0 = blockIdx.y * 32 + wm * 16;
    const int n0 = blockIdx.x * 64 + wn * 16;

    const __bf16* Ar = A + (size_t)(m0 + nn) * K;
    const __bf16* Wr = W + (size_t)(n0 + nn) * K;

    v8f acc = {0.f, 0.f, 0.f, 0.f, 0.f, 0.f, 0.f, 0.f};
    for (int k0 = 0; k0 < K; k0 += 32) {
        FragBf a, bf;
        const int ka = k0 + 8 * h;
        a.f4[0] = *(const float4*)(Ar + ka);
        a.f4[1] = *(const float4*)(Ar + ka + 16);
        const int kb = k0 + 8 * h;
        bf.f4[0] = *(const float4*)(Wr + kb);
        bf.f4[1] = *(const float4*)(Wr + kb + 16);
        acc = __builtin_amdgcn_wmma_f32_16x16x32_bf16(
                false, a.v, false, bf.v, (short)0, acc, false, false);
    }

    const int n = n0 + nn;
    const float bs = bias[n];
    const float s1 = rsqrtf(v1[n] + EPS) * g1[n];
    const float c1 = be1[n] - m1[n] * s1;
    float s2 = 0.f, c2 = 0.f;
    if (mode == 0) { s2 = rsqrtf(v2[n] + EPS) * g2[n]; c2 = be2[n] - m2[n] * s2; }

    for (int v = 0; v < 8; ++v) {
        int m = m0 + v + 8 * h;
        float t = (acc[v] + bs) * s1 + c1;
        t = fmaxf(t, 0.f);
        if (mode == 2) {
            int k = m >> 8, bidx = m & 255, j = k * 128 + n;
            float ss = rsqrtf(v2[j] + EPS) * g2[j];
            t = (t - m2[j]) * ss + be2[j];
            out[(size_t)bidx * 512 + j] = (__bf16)t;
        } else {
            if (mode == 0) t = t * s2 + c2;
            out[(size_t)m * N + n] = (__bf16)t;
        }
    }
}

// ---------------------------------------------------------------------------
// Final 51-class heads (27 MFLOP total -- negligible; plain VALU dot products)
// d_out = [x_prelogits (256x51) | y_prelogits (256x51)] f32
// ---------------------------------------------------------------------------
__global__ void logits_kernel(const __bf16* __restrict__ xg, const __bf16* __restrict__ Y,
                              const float* __restrict__ Wcg, const float* __restrict__ bcg,
                              const float* __restrict__ Wck, const float* __restrict__ bck,
                              float* __restrict__ out) {
    int idx = blockIdx.x * blockDim.x + threadIdx.x;
    const int total = 2 * 256 * 51;
    if (idx >= total) return;
    int which = idx / (256 * 51);
    int r = idx - which * (256 * 51);
    int b = r / 51, j = r - b * 51;
    const __bf16* a  = (which == 0 ? xg : Y) + (size_t)b * 512;
    const float*  wv = (which == 0 ? Wcg : Wck) + (size_t)j * 512;
    float s = (which == 0 ? bcg : bck)[j];
    for (int nidx = 0; nidx < 512; ++nidx) s += (float)a[nidx] * wv[nidx];
    out[idx] = s;
}

// ---------------------------------------------------------------------------
extern "C" void kernel_launch(void* const* d_in, const int* in_sizes, int n_in,
                              void* d_out, int out_size, void* d_ws, size_t ws_size,
                              hipStream_t stream) {
    const float* x        = (const float*)d_in[0];
    const float* Wg       = (const float*)d_in[1];
    const float* bg       = (const float*)d_in[2];
    const float* g_gamma  = (const float*)d_in[3];
    const float* g_beta   = (const float*)d_in[4];
    const float* g_mean   = (const float*)d_in[5];
    const float* g_var    = (const float*)d_in[6];
    const float* gb_gamma = (const float*)d_in[7];
    const float* gb_beta  = (const float*)d_in[8];
    const float* gb_mean  = (const float*)d_in[9];
    const float* gb_var   = (const float*)d_in[10];
    const float* Wcg      = (const float*)d_in[11];
    const float* bcg      = (const float*)d_in[12];
    const float* W1       = (const float*)d_in[13];
    const float* b1       = (const float*)d_in[14];
    const float* n1_gamma = (const float*)d_in[15];
    const float* n1_beta  = (const float*)d_in[16];
    const float* n1_mean  = (const float*)d_in[17];
    const float* n1_var   = (const float*)d_in[18];
    const float* W2       = (const float*)d_in[19];
    const float* b2       = (const float*)d_in[20];
    const float* n2_gamma = (const float*)d_in[21];
    const float* n2_beta  = (const float*)d_in[22];
    const float* n2_mean  = (const float*)d_in[23];
    const float* n2_var   = (const float*)d_in[24];
    const float* bk_gamma = (const float*)d_in[25];   // [4,128]
    const float* bk_beta  = (const float*)d_in[26];
    const float* bk_mean  = (const float*)d_in[27];
    const float* bk_var   = (const float*)d_in[28];
    const float* Wck      = (const float*)d_in[29];
    const float* bck      = (const float*)d_in[30];
    float* out = (float*)d_out;

    // workspace carve
    char* ws = (char*)d_ws;
    size_t off = 0;
    auto carve = [&](size_t bytes) -> void* {
        void* p = ws + off;
        off = (off + bytes + 255) & ~((size_t)255);
        return p;
    };
    float*  pooled   = (float*)carve((size_t)256 * 2048 * 4);
    __bf16* pooledBf = (__bf16*)carve((size_t)256 * 2048 * 2);
    __bf16* PkAll    = (__bf16*)carve((size_t)1024 * 2048 * 2);
    __bf16* WgBf     = (__bf16*)carve((size_t)512 * 2048 * 2);
    __bf16* W1Bf     = (__bf16*)carve((size_t)512 * 2048 * 2);
    __bf16* W2Bf     = (__bf16*)carve((size_t)128 * 512 * 2);
    __bf16* xgBf     = (__bf16*)carve((size_t)256 * 512 * 2);
    __bf16* H1Bf     = (__bf16*)carve((size_t)1024 * 512 * 2);
    __bf16* YBf      = (__bf16*)carve((size_t)256 * 512 * 2);

    // 1) weights -> bf16
    cvt_bf16_kernel<<<(512 * 2048 + 255) / 256, 256, 0, stream>>>(Wg, WgBf, 512 * 2048);
    cvt_bf16_kernel<<<(512 * 2048 + 255) / 256, 256, 0, stream>>>(W1, W1Bf, 512 * 2048);
    cvt_bf16_kernel<<<(128 * 512 + 255) / 256, 256, 0, stream>>>(W2, W2Bf, 128 * 512);

    // 2) pooled (B*C = 524288 channels, 8 per block)
    pool_kernel<<<524288 / 8, 256, 0, stream>>>(x, pooled, pooledBf);

    // 3) per-sample k-means + masked-pooled Pk
    kmeans_kernel<<<256, 256, 0, stream>>>(x, pooled, PkAll);

    // 4) global branch: xg = BN_gb(relu(BN_g(pooled@Wg^T + bg)))   [256,512]
    gemm_bf16_kernel<<<dim3(512 / 64, 256 / 32), 256, 0, stream>>>(
        pooledBf, WgBf, 256, 512, 2048, 0, bg,
        g_gamma, g_beta, g_mean, g_var,
        gb_gamma, gb_beta, gb_mean, gb_var, xgBf);

    // 5) H1 = relu(BN_n1(Pk@W1^T + b1))   [1024,512]
    gemm_bf16_kernel<<<dim3(512 / 64, 1024 / 32), 256, 0, stream>>>(
        PkAll, W1Bf, 1024, 512, 2048, 1, b1,
        n1_gamma, n1_beta, n1_mean, n1_var,
        nullptr, nullptr, nullptr, nullptr, H1Bf);

    // 6) H2 = BN_bk[k](relu(BN_n2(H1@W2^T + b2))) -> concat Y [256,512]
    gemm_bf16_kernel<<<dim3(128 / 64, 1024 / 32), 256, 0, stream>>>(
        H1Bf, W2Bf, 1024, 128, 512, 2, b2,
        n2_gamma, n2_beta, n2_mean, n2_var,
        bk_gamma, bk_beta, bk_mean, bk_var, YBf);

    // 7) heads
    logits_kernel<<<(2 * 256 * 51 + 255) / 256, 256, 0, stream>>>(
        xgBf, YBf, Wcg, bcg, Wck, bck, out);

    (void)in_sizes; (void)n_in; (void)out_size; (void)ws_size;
}
